// Net_996432413190
// MI455X (gfx1250) — matrix operand order repaired and checked by reading, not compiled, not run on hardware
//
#include <hip/hip_runtime.h>

#define NN 100000
#define EE 1200000
#define HD 64

typedef __attribute__((ext_vector_type(16))) __bf16 v16bf;
typedef __attribute__((ext_vector_type(8)))  float  v8f;

struct U8 { unsigned int v[8]; };

__device__ __forceinline__ unsigned short f2bf(float f) {
    unsigned int u = __float_as_uint(f);
    unsigned int r = u + 0x7FFFu + ((u >> 16) & 1u);   // round-to-nearest-even
    return (unsigned short)(r >> 16);
}
__device__ __forceinline__ unsigned int pack2(float a, float b) {
    return (unsigned int)f2bf(a) | ((unsigned int)f2bf(b) << 16);
}
__device__ __forceinline__ float elu1(float y) {
    return y > 0.f ? y : (__expf(y) - 1.f);
}

// ---------------- encoder: 15 -> 64 (ELU) -> 64 (ELU), 4 nodes / block ----------------
__global__ void encoder_kernel(const float* __restrict__ x,
                               const float* __restrict__ w1, const float* __restrict__ b1,
                               const float* __restrict__ w2, const float* __restrict__ b2,
                               float* __restrict__ out) {
    __shared__ float sW1[15 * 64];
    __shared__ float sW2[64 * 64];
    __shared__ float sB1[64], sB2[64];
    __shared__ float sX[4 * 15];
    __shared__ float sT[4 * 64];
    int t = threadIdx.x;
    for (int i = t; i < 15 * 64; i += 256) sW1[i] = w1[i];
    for (int i = t; i < 64 * 64; i += 256) sW2[i] = w2[i];
    if (t < 64) { sB1[t] = b1[t]; sB2[t] = b2[t]; }
    if (t < 60) {
        int nn = blockIdx.x * 4 + t / 15;
        sX[t] = (nn < NN) ? x[nn * 15 + (t % 15)] : 0.f;
    }
    __syncthreads();
    int nl = t >> 6, j = t & 63;
    float acc = sB1[j];
    #pragma unroll
    for (int k = 0; k < 15; ++k) acc += sX[nl * 15 + k] * sW1[k * 64 + j];
    sT[nl * 64 + j] = elu1(acc);
    __syncthreads();
    float acc2 = sB2[j];
    #pragma unroll 8
    for (int k = 0; k < 64; ++k) acc2 += sT[nl * 64 + k] * sW2[k * 64 + j];
    int node = blockIdx.x * 4 + nl;
    if (node < NN) out[node * 64 + j] = elu1(acc2);
}

// ---------------- in-degree (same dst for all 3 convs) ----------------
__global__ void degree_kernel(const int* __restrict__ dst, float* __restrict__ deg) {
    int e = blockIdx.x * 256 + threadIdx.x;
    if (e < EE) atomicAdd(&deg[dst[e]], 1.f);
}

// ---------------- EdgeConv: (E x 128) @ (128 x 64) via bf16 WMMA + ELU + BN + scatter ----------------
// sum has NN+1 rows; row NN is a dump row for (never-taken, E%16==0) inactive tile slots,
// keeping the 32-atomic epilogue completely branchless (no exec-mask juggling).
__global__ void __launch_bounds__(256) edgeconv_kernel(
        const float* __restrict__ xin, const int* __restrict__ src, const int* __restrict__ dst,
        const float* __restrict__ W, const float* __restrict__ bias,
        const float* __restrict__ gamma, const float* __restrict__ beta,
        const float* __restrict__ mean, const float* __restrict__ var,
        float* __restrict__ sum) {
    __shared__ unsigned int sW[128 * 32];      // 128x64 bf16 weights (u32 = 2 bf16), stored N x K, k-pairs contiguous per column n
    __shared__ unsigned int sMsg[8 * 16 * 64]; // per wave: 16x128 bf16 message tile (wave-private)
    __shared__ float sBias[64], sScale[64], sShift[64];
    __shared__ int   sDst[8 * 16];             // wave-private
    int t = threadIdx.x;

    for (int i = t; i < 4096; i += 256) sW[i] = pack2(W[(2 * (i & 63)) * 64 + (i >> 6)], W[(2 * (i & 63) + 1) * 64 + (i >> 6)]);
    if (t < 64) {
        float sc = gamma[t] * rsqrtf(var[t] + 1e-5f);
        sScale[t] = sc; sShift[t] = beta[t] - mean[t] * sc; sBias[t] = bias[t];
    }
    __syncthreads();   // sW/sBias/sScale/sShift are read by all waves below

    int lane = t & 31, wave = t >> 5;
    int totalTiles = (EE + 15) / 16;
    int wavesTotal = gridDim.x * 8;
    unsigned int* myMsg = sMsg + wave * 1024;
    int* myDst = sDst + wave * 16;
    int m16 = lane & 15, hi = lane >> 4;

    // Wave-independent grid-stride loop. No barriers needed inside: sMsg/sDst are
    // wave-private and DS ops are in-order within a wave (RAW and WAR both safe).
    for (int tile = blockIdx.x * 8 + wave; tile < totalTiles; tile += wavesTotal) {
        // ---- gather + build message tile: msg[m][0:64]=x_dst, msg[m][64:128]=x_src-x_dst ----
        {
            int m = lane >> 1, half = lane & 1;
            int e = tile * 16 + m;
            bool act = (e < EE);
            int eid = act ? e : 0;
            int sN = src[eid], dN = dst[eid];
            myDst[m] = act ? dN : NN;          // both lanes of an edge write the same value
            const float4* xd = (const float4*)(xin + (size_t)dN * 64 + half * 32);
            const float4* xs = (const float4*)(xin + (size_t)sN * 64 + half * 32);
            unsigned int* rowA = myMsg + m * 64 + half * 16;       // x_dst part
            unsigned int* rowB = myMsg + m * 64 + 32 + half * 16;  // x_src - x_dst part
            #pragma unroll
            for (int j = 0; j < 8; ++j) {
                float4 a = xd[j];
                float4 b = xs[j];
                rowA[2 * j]     = pack2(a.x, a.y);
                rowA[2 * j + 1] = pack2(a.z, a.w);
                rowB[2 * j]     = pack2(b.x - a.x, b.y - a.y);
                rowB[2 * j + 1] = pack2(b.z - a.z, b.w - a.w);
            }
        }
        // ---- destinations for this lane's 8 output rows ----
        int dreg[8];
        #pragma unroll
        for (int r = 0; r < 8; ++r) dreg[r] = myDst[r + (hi << 3)];
        // ---- 16x128 @ 128x64 = 16x64, as 4 k-tiles x 4 n-tiles of v_wmma_f32_16x16x32_bf16 ----
        v8f cacc[4] = {};
        #pragma unroll
        for (int kt = 0; kt < 4; ++kt) {
            U8 au;
            #pragma unroll
            for (int v = 0; v < 8; ++v)  // ISA A layout: k-pair = (v>=4)*16 + hi*8 + (v&3)*2
                au.v[v] = myMsg[m16 * 64 + kt * 16 + ((v >> 2) << 3) + (hi << 2) + (v & 3)];
            v16bf a = __builtin_bit_cast(v16bf, au);
            #pragma unroll
            for (int nt = 0; nt < 4; ++nt) {
                U8 bu;
                #pragma unroll
                for (int j = 0; j < 8; ++j)  // ISA B layout: lane = N col, k-pair = (j>=4)*16 + hi*8 + (j&3)*2
                    bu.v[j] = sW[(nt * 16 + m16) * 64 + kt * 16 + ((j >> 2) << 3) + (hi << 2) + (j & 3)];
                v16bf b = __builtin_bit_cast(v16bf, bu);
                cacc[nt] = __builtin_amdgcn_wmma_f32_16x16x32_bf16(
                    false, a, false, b, (short)0, cacc[nt], false, false);
            }
        }
        // ---- epilogue: bias + ELU + BN, branchless scatter-add to destination nodes ----
        #pragma unroll
        for (int nt = 0; nt < 4; ++nt) {
            int n = nt * 16 + m16;
            float bi = sBias[n], sc = sScale[n], sh = sShift[n];
            #pragma unroll
            for (int r = 0; r < 8; ++r) {
                float y = cacc[nt][r] + bi;    // C layout: vgpr r, lane half -> row m = r + hi*8
                y = elu1(y);
                y = y * sc + sh;
                atomicAdd(&sum[(size_t)dreg[r] * 64 + n], y);
            }
        }
    }
}

// ---------------- mean + optional residual add ----------------
__global__ void finalize_kernel(const float* __restrict__ sum, const float* __restrict__ deg,
                                float* __restrict__ outF, const float* __restrict__ addSrc,
                                float* __restrict__ outG) {
    int i = blockIdx.x * 256 + threadIdx.x;
    if (i >= NN * 64) return;
    float c = deg[i >> 6];
    float f = sum[i] / fmaxf(c, 1.f);
    if (outF) outF[i] = f;
    if (outG) outG[i] = f + addSrc[i];
}

// ---------------- head: 64 -> 32 (ELU) -> 16 (ELU) -> 8 ----------------
__global__ void head_kernel(const float* __restrict__ fin,
                            const float* __restrict__ w1, const float* __restrict__ b1,
                            const float* __restrict__ w2, const float* __restrict__ b2,
                            const float* __restrict__ w3, const float* __restrict__ b3,
                            float* __restrict__ out) {
    __shared__ float sW1[64 * 32], sW2[32 * 16], sW3[16 * 8];
    __shared__ float sB1[32], sB2[16], sB3[8];
    int t = threadIdx.x;
    for (int i = t; i < 64 * 32; i += 256) sW1[i] = w1[i];
    for (int i = t; i < 32 * 16; i += 256) sW2[i] = w2[i];
    for (int i = t; i < 16 * 8;  i += 256) sW3[i] = w3[i];
    if (t < 32) sB1[t] = b1[t];
    if (t < 16) sB2[t] = b2[t];
    if (t < 8)  sB3[t] = b3[t];
    __syncthreads();
    int node = blockIdx.x * 256 + t;
    if (node >= NN) return;
    float f[64];
    const float4* fp = (const float4*)(fin + (size_t)node * 64);
    #pragma unroll
    for (int k = 0; k < 16; ++k) { float4 q = fp[k]; f[4*k]=q.x; f[4*k+1]=q.y; f[4*k+2]=q.z; f[4*k+3]=q.w; }
    float o1[32];
    #pragma unroll
    for (int j = 0; j < 32; ++j) {
        float a = sB1[j];
        for (int k = 0; k < 64; ++k) a += f[k] * sW1[k * 32 + j];
        o1[j] = elu1(a);
    }
    float o2[16];
    #pragma unroll
    for (int j = 0; j < 16; ++j) {
        float a = sB2[j];
        for (int k = 0; k < 32; ++k) a += o1[k] * sW2[k * 16 + j];
        o2[j] = elu1(a);
    }
    #pragma unroll
    for (int j = 0; j < 8; ++j) {
        float a = sB3[j];
        for (int k = 0; k < 16; ++k) a += o2[k] * sW3[k * 8 + j];
        out[(size_t)node * 8 + j] = a;
    }
}

__global__ void batch_copy_kernel(const int* __restrict__ batch, float* __restrict__ out) {
    int i = blockIdx.x * 256 + threadIdx.x;
    if (i < NN) out[i] = (float)batch[i];
}

extern "C" void kernel_launch(void* const* d_in, const int* in_sizes, int n_in,
                              void* d_out, int out_size, void* d_ws, size_t ws_size,
                              hipStream_t stream) {
    (void)in_sizes; (void)n_in; (void)out_size; (void)ws_size;
    const float* x      = (const float*)d_in[0];
    const int*   ei     = (const int*)  d_in[1];
    const int*   batch  = (const int*)  d_in[2];
    const float* enc_w1 = (const float*)d_in[3];
    const float* enc_b1 = (const float*)d_in[4];
    const float* enc_w2 = (const float*)d_in[5];
    const float* enc_b2 = (const float*)d_in[6];
    const float *cw[3], *cb[3], *cg[3], *cbe[3], *cm[3], *cv[3];
    for (int i = 0; i < 3; ++i) {
        cw[i]  = (const float*)d_in[7 + 6 * i + 0];
        cb[i]  = (const float*)d_in[7 + 6 * i + 1];
        cg[i]  = (const float*)d_in[7 + 6 * i + 2];
        cbe[i] = (const float*)d_in[7 + 6 * i + 3];
        cm[i]  = (const float*)d_in[7 + 6 * i + 4];
        cv[i]  = (const float*)d_in[7 + 6 * i + 5];
    }
    const float* out_w1 = (const float*)d_in[25];
    const float* out_b1 = (const float*)d_in[26];
    const float* out_w2 = (const float*)d_in[27];
    const float* out_b2 = (const float*)d_in[28];
    const float* out_w3 = (const float*)d_in[29];
    const float* out_b3 = (const float*)d_in[30];

    char* ws = (char*)d_ws;
    size_t S  = (size_t)NN * 64 * sizeof(float);         // feature buffer
    size_t S2 = (size_t)(NN + 1) * 64 * sizeof(float);   // accumulator (+dump row)
    float* B0  = (float*)(ws);               // h, later g3
    float* B1  = (float*)(ws + S);           // f1, later f3
    float* B2  = (float*)(ws + 2 * S);       // g2
    float* SUM = (float*)(ws + 3 * S);
    float* DEG = (float*)(ws + 3 * S + S2);
    const int* src = ei;
    const int* dst = ei + EE;

    encoder_kernel<<<(NN + 3) / 4, 256, 0, stream>>>(x, enc_w1, enc_b1, enc_w2, enc_b2, B0);
    hipMemsetAsync(DEG, 0, NN * sizeof(float), stream);
    degree_kernel<<<(EE + 255) / 256, 256, 0, stream>>>(dst, DEG);

    const int CB = 1024;
    const int FG = (NN * 64 + 255) / 256;
    // conv1: in=h -> f1 (B1); g2 = f1 + h -> B2
    hipMemsetAsync(SUM, 0, S2, stream);
    edgeconv_kernel<<<CB, 256, 0, stream>>>(B0, src, dst, cw[0], cb[0], cg[0], cbe[0], cm[0], cv[0], SUM);
    finalize_kernel<<<FG, 256, 0, stream>>>(SUM, DEG, B1, B0, B2);
    // conv2: in=g2 -> f2; g3 = f2 + f1 -> B0
    hipMemsetAsync(SUM, 0, S2, stream);
    edgeconv_kernel<<<CB, 256, 0, stream>>>(B2, src, dst, cw[1], cb[1], cg[1], cbe[1], cm[1], cv[1], SUM);
    finalize_kernel<<<FG, 256, 0, stream>>>(SUM, DEG, nullptr, B1, B0);
    // conv3: in=g3 -> f3 (B1)
    hipMemsetAsync(SUM, 0, S2, stream);
    edgeconv_kernel<<<CB, 256, 0, stream>>>(B0, src, dst, cw[2], cb[2], cg[2], cbe[2], cm[2], cv[2], SUM);
    finalize_kernel<<<FG, 256, 0, stream>>>(SUM, DEG, B1, nullptr, nullptr);

    head_kernel<<<(NN + 255) / 256, 256, 0, stream>>>(B1, out_w1, out_b1, out_w2, out_b2, out_w3, out_b3,
                                                      (float*)d_out);
    batch_copy_kernel<<<(NN + 255) / 256, 256, 0, stream>>>(batch, (float*)d_out + (size_t)NN * 8);
}
